// PathEmbedding_89197880803799
// MI455X (gfx1250) — compile-verified, matrix operand order checked by reading, not run on hardware
//
#include <hip/hip_runtime.h>

typedef __attribute__((ext_vector_type(16))) _Float16 v16h;
typedef __attribute__((ext_vector_type(4)))  _Float16 v4h;
typedef __attribute__((ext_vector_type(8)))  float    v8f;
typedef __attribute__((ext_vector_type(4)))  float    v4f;

#define N_NODE  50000
#define LPATH   6
#define D_NODE  64
#define DPATH   128
#define DF      320      // D_NODE*(LPATH-1)
#define NCHUNK  10       // DF / 32  (K-chunks per wmma)
#define NTILES  8        // DPATH / 16
#define MAXGRID 1024

// 256 threads = 8 wave32; each wave computes 16-row (n,k) tiles x all 128 outputs.
// Grid-stride over block-tiles so the LDS-staged W is reused across several tiles.
__global__ __launch_bounds__(256)
void path_embed_wmma(const float* __restrict__ feat,
                     const float* __restrict__ weight,
                     const float* __restrict__ W,
                     const long long* __restrict__ paths,
                     const long long* __restrict__ edges,
                     float* __restrict__ out,
                     int nk_rows, int n_block_tiles)
{
    // B fragments of W in f16, pre-swizzled: [NTILES*NCHUNK][lane 0..31][16 halves] = 80 KB
    extern __shared__ _Float16 smem[];

    const int tid = threadIdx.x;

    // ---- cooperative f32->f16 convert of W into WMMA B-fragment layout (vectorized) ----
    // Each aligned group of 4 halves maps to 4 consecutive W elements:
    //   h4=0 -> kloc base+0..3, h4=1 -> base+4..7, h4=2 -> base+16..19, h4=3 -> base+20..23
    for (int idx4 = tid; idx4 < NTILES * NCHUNK * 32 * 4; idx4 += 256) {
        int h4   = idx4 & 3;
        int lane = (idx4 >> 2) & 31;
        int frag = idx4 >> 7;
        int c    = frag % NCHUNK;
        int nt   = frag / NCHUNK;
        int nloc = lane & 15;
        int base = (lane & 16) ? 8 : 0;
        int kloc = base + ((h4 & 2) ? 8 : 0) + h4 * 4;
        v4f w4 = *(const v4f*)&W[(nt * 16 + nloc) * DF + c * 32 + kloc];
        v4h h  = { (_Float16)w4.x, (_Float16)w4.y, (_Float16)w4.z, (_Float16)w4.w };
        *(v4h*)&smem[idx4 * 4] = h;
    }
    __syncthreads();

    const int wave  = tid >> 5;
    const int lane  = tid & 31;
    const int mrow  = lane & 15;                 // A-matrix: lane L -> M = L%16
    const int kbase = (lane & 16) ? 8 : 0;       // A layout K offset per lane half
    const int moff  = (lane & 16) ? 8 : 0;       // C/D layout M offset per lane half
    const int ncol  = lane & 15;

    for (int bt = blockIdx.x; bt < n_block_tiles; bt += gridDim.x) {
        const int rowbase = (bt * 8 + wave) * 16;
        if (rowbase >= nk_rows) continue;        // uniform across the wave
        const int row = rowbase + mrow;

        // ---- per-row path state: clamped feat row pointers + zero-pad masks ----
        const long long* prow = paths + (long long)row * LPATH;
        const long long* erow = edges + (long long)row * (LPATH - 1);
        const float* fptr[LPATH];
        float        fmask[LPATH];
#pragma unroll
        for (int l = 0; l < LPATH; ++l) {
            long long idx = __builtin_nontemporal_load(&prow[l]);  // read-once stream
            bool ok = (idx < (long long)N_NODE);                   // N_NODE => zero-pad row
            fptr[l]  = feat + (ok ? idx : 0) * D_NODE;
            fmask[l] = ok ? 1.0f : 0.0f;
        }
        float ew[LPATH - 1];
#pragma unroll
        for (int l = 0; l < LPATH - 1; ++l) {
            long long e = __builtin_nontemporal_load(&erow[l]);
            ew[l] = weight[e] * fmask[l + 1];    // fold right-side pad mask into weight
        }

        // ---- phase 1: build ALL A fragments (batches all gathers up front) ----
        v16h a[NCHUNK];
#pragma unroll
        for (int c = 0; c < NCHUNK; ++c) {
#pragma unroll
            for (int r = 0; r < 2; ++r) {        // two runs of 8 consecutive f-values
                int f0 = c * 32 + kbase + r * 16;
                int l  = f0 >> 6;                // which path segment
                int d2 = f0 & 63;                // offset in feat row (multiple of 8)
                v4f x0 = *(const v4f*)(fptr[l]     + d2);
                v4f x1 = *(const v4f*)(fptr[l]     + d2 + 4);
                v4f y0 = *(const v4f*)(fptr[l + 1] + d2);
                v4f y1 = *(const v4f*)(fptr[l + 1] + d2 + 4);
                float ml = fmask[l], wl = ew[l];
#pragma unroll
                for (int j = 0; j < 4; ++j) {
                    a[c][r * 8 + j]     = (_Float16)(ml * x0[j] + wl * y0[j]);
                    a[c][r * 8 + 4 + j] = (_Float16)(ml * x1[j] + wl * y1[j]);
                }
            }
        }

        // ---- phase 2: N-tile-outer wmma sweep; single accumulator per tile ----
        // All 10 B loads of a tile are independent of its wmmas -> scheduler can
        // hoist the ds_load_b128 clause and count the dscnt waits down.
#pragma unroll
        for (int nt = 0; nt < NTILES; ++nt) {
            v8f acc = {0.f, 0.f, 0.f, 0.f, 0.f, 0.f, 0.f, 0.f};
#pragma unroll
            for (int c = 0; c < NCHUNK; ++c) {
                v16h b = *(const v16h*)(smem + ((nt * NCHUNK + c) * 32 + lane) * 16);
                acc = __builtin_amdgcn_wmma_f32_16x16x32_f16(
                    false, a[c], false, b, (short)0, acc, false, false);
            }
            // store this 16x16 tile (non-temporal: write-once stream, keep L2 for feat)
            // C/D layout: lanes 0-15: M=v, N=lane; lanes 16-31: M=8+v, N=lane-16
#pragma unroll
            for (int v = 0; v < 8; ++v) {
                __builtin_nontemporal_store(
                    acc[v],
                    &out[(long long)(rowbase + moff + v) * DPATH + nt * 16 + ncol]);
            }
        }
    }
}

extern "C" void kernel_launch(void* const* d_in, const int* in_sizes, int n_in,
                              void* d_out, int out_size, void* d_ws, size_t ws_size,
                              hipStream_t stream) {
    const float*     feat   = (const float*)d_in[0];
    const float*     weight = (const float*)d_in[1];
    const float*     W      = (const float*)d_in[2];
    const long long* paths  = (const long long*)d_in[3];
    const long long* edges  = (const long long*)d_in[4];
    float* out = (float*)d_out;

    int nk_rows = out_size / DPATH;                  // 400000
    int waves   = (nk_rows + 15) / 16;               // 25000
    int tiles   = (waves + 7) / 8;                   // 3125 block-tiles
    int grid    = tiles < MAXGRID ? tiles : MAXGRID; // grid-stride amortizes W staging
    size_t shmem = (size_t)NTILES * NCHUNK * 32 * 16 * sizeof(_Float16);  // 81920 B

    hipFuncSetAttribute((const void*)path_embed_wmma,
                        hipFuncAttributeMaxDynamicSharedMemorySize, (int)shmem);
    path_embed_wmma<<<grid, 256, shmem, stream>>>(feat, weight, W, paths, edges,
                                                  out, nk_rows, tiles);
}
